// Net_19679540150408
// MI455X (gfx1250) — compile-verified, matrix operand order checked by reading, not run on hardware
//
#include <hip/hip_runtime.h>
#include <hip/hip_bf16.h>

// ---------------- problem constants ----------------
#define NG 50000
#define ND 20000
#define EG 300000
#define ED 150000
#define HF 256            // feature width (FG=FD=H=256)
#define BB 65536          // batch pairs
#define CHUNK 16384       // MLP row chunk (4 chunks)

typedef __attribute__((ext_vector_type(16))) _Float16 v16h;
typedef __attribute__((ext_vector_type(8)))  _Float16 v8h;
typedef __attribute__((ext_vector_type(4)))  _Float16 v4h;
typedef __attribute__((ext_vector_type(8)))  float    v8f;

// ---------------- utility kernels ----------------
__global__ void zero_kernel(float* __restrict__ p, long n) {
    long i = (long)blockIdx.x * blockDim.x + threadIdx.x;
    if (i < n) p[i] = 0.0f;
}

// one block (256 threads) per edge: agg[dst] += x[src]
__global__ void scatter_add_kernel(const float* __restrict__ x,
                                   const int* __restrict__ ei, int E,
                                   float* __restrict__ agg) {
    int e = blockIdx.x;
    int f = threadIdx.x;
    int src = ei[e];
    int dst = ei[E + e];
    atomicAdd(&agg[(size_t)dst * HF + f], x[(size_t)src * HF + f]);
}

// ---------------- fused WMMA GEMM ----------------
// C[M,N] = act( A1[M,K1]@W1[K1,N] + A2[M,K2]@W2[K2,N] + bias ),
// optional per-column sum / sumsq accumulation (for BatchNorm stats).
#define BM 128
#define BN 64
#define BK 64
#define BSTRIDE (BK + 8)   // 72 halfs = 144 B row stride (16B-aligned runs)

// Each lane's 16 fragment halves = two contiguous 16-byte runs (ISA 7.12.2).
__device__ __forceinline__ v16h frag_load(const _Float16* lo, const _Float16* hi) {
    v8h a = *(const v8h*)lo;
    v8h b = *(const v8h*)hi;
    return __builtin_shufflevector(a, b, 0, 1, 2, 3, 4, 5, 6, 7,
                                         8, 9, 10, 11, 12, 13, 14, 15);
}

__global__ __launch_bounds__(256)
void wmma_gemm_kernel(const float* __restrict__ A1, const float* __restrict__ W1,
                      const float* __restrict__ A2, const float* __restrict__ W2,
                      const float* __restrict__ bias, float* __restrict__ C,
                      int M, int N, int K1, int K2, int doLrelu,
                      float* __restrict__ colsum, float* __restrict__ colsumsq) {
    __shared__ _Float16 As[BM][BK];            // [m][k]  16 KB, 128 B rows
    __shared__ _Float16 BsT[BN][BSTRIDE];      // [n][k]   9 KB, 144 B rows

    const int tid  = threadIdx.x;
    const int lane = tid & 31;
    const int wave = tid >> 5;
    const int half = lane >> 4;      // 0/1
    const int l16  = lane & 15;
    const int wm   = wave >> 1;      // 0..3 -> 32-row strip
    const int wn   = wave & 1;       // 0..1 -> 32-col strip
    const int m0   = blockIdx.x * BM;
    const int n0   = blockIdx.y * BN;

    v8f acc00 = {}, acc01 = {}, acc10 = {}, acc11 = {};

    for (int srcSel = 0; srcSel < 2; ++srcSel) {
        const float* A = (srcSel == 0) ? A1 : A2;
        const float* W = (srcSel == 0) ? W1 : W2;
        const int    K = (srcSel == 0) ? K1 : K2;
        if (A == nullptr || K == 0) continue;

        for (int kb = 0; kb < K; kb += BK) {
            if (kb + BK < K)
                __builtin_prefetch(&A[(size_t)m0 * K + kb + BK], 0, 1);

            // ---- stage A tile: 128x64 f32 -> f16, float4 loads, b64 LDS stores ----
            // 8192 floats = 2048 float4 groups; 8 groups per thread, coalesced.
#pragma unroll
            for (int i = 0; i < 8; ++i) {
                int g = tid + i * 256;
                int r = g >> 4;                 // 16 float4-groups per 64-wide row
                int c = (g & 15) << 2;
                int gm = m0 + r;
                int rc = (gm < M) ? gm : (M - 1);       // clamp: address always valid
                float msk = (gm < M) ? 1.0f : 0.0f;     // branch-free zero fill
                const float4 v = *(const float4*)&A[(size_t)rc * K + kb + c];
                v4h hv;
                hv[0] = (_Float16)(v.x * msk);
                hv[1] = (_Float16)(v.y * msk);
                hv[2] = (_Float16)(v.z * msk);
                hv[3] = (_Float16)(v.w * msk);
                *(v4h*)&As[r][c] = hv;                  // 8B aligned
            }
            // ---- stage W tile transposed: 64x64, float4 loads, b16 stores ----
            // 4096 floats = 1024 float4 groups; 4 per thread.
#pragma unroll
            for (int i = 0; i < 4; ++i) {
                int g = tid + i * 256;
                int r = g >> 4;                 // k row 0..63
                int c = (g & 15) << 2;          // n col 0..63 (x4)
                const float4 v = *(const float4*)&W[(size_t)(kb + r) * N + n0 + c];
                BsT[c + 0][r] = (_Float16)v.x;
                BsT[c + 1][r] = (_Float16)v.y;
                BsT[c + 2][r] = (_Float16)v.z;
                BsT[c + 3][r] = (_Float16)v.w;
            }
            __syncthreads();

            // ---- two K=32 WMMA steps per staged tile ----
#pragma unroll
            for (int ks = 0; ks < BK; ks += 32) {
                const int klo = ks + 8 * half;        // first 16-byte run
                const int khi = klo + 16;             // second 16-byte run
                const int rowA0 = wm * 32 + l16;
                const int rowA1 = rowA0 + 16;
                const int colB0 = wn * 32 + l16;
                const int colB1 = colB0 + 16;

                v16h a0 = frag_load(&As[rowA0][klo],  &As[rowA0][khi]);
                v16h a1 = frag_load(&As[rowA1][klo],  &As[rowA1][khi]);
                v16h b0 = frag_load(&BsT[colB0][klo], &BsT[colB0][khi]);
                v16h b1 = frag_load(&BsT[colB1][klo], &BsT[colB1][khi]);

                acc00 = __builtin_amdgcn_wmma_f32_16x16x32_f16(
                    false, a0, false, b0, (short)0, acc00, false, false);
                acc01 = __builtin_amdgcn_wmma_f32_16x16x32_f16(
                    false, a0, false, b1, (short)0, acc01, false, false);
                acc10 = __builtin_amdgcn_wmma_f32_16x16x32_f16(
                    false, a1, false, b0, (short)0, acc10, false, false);
                acc11 = __builtin_amdgcn_wmma_f32_16x16x32_f16(
                    false, a1, false, b1, (short)0, acc11, false, false);
            }
            __syncthreads();
        }
    }

    // ---- epilogue: bias, leaky-relu, store, BN column stats ----
    const int colA = n0 + wn * 32 + l16;
    const int colB = colA + 16;
    const float bA = bias ? bias[colA] : 0.0f;
    const float bB = bias ? bias[colB] : 0.0f;
    float sA = 0.f, qA = 0.f, sB = 0.f, qB = 0.f;
#pragma unroll
    for (int r = 0; r < 2; ++r) {
        const int row_base = m0 + wm * 32 + r * 16 + 8 * half;
        v8f accA = r ? acc10 : acc00;
        v8f accB = r ? acc11 : acc01;
#pragma unroll
        for (int v = 0; v < 8; ++v) {
            int row = row_base + v;
            if (row < M) {
                float zA = accA[v] + bA;
                float zB = accB[v] + bB;
                if (doLrelu) {
                    zA = (zA >= 0.f) ? zA : 0.01f * zA;
                    zB = (zB >= 0.f) ? zB : 0.01f * zB;
                }
                C[(size_t)row * N + colA] = zA;
                C[(size_t)row * N + colB] = zB;
                sA += zA; qA += zA * zA;
                sB += zB; qB += zB * zB;
            }
        }
    }
    if (colsum) {
        atomicAdd(&colsum[colA],   sA);
        atomicAdd(&colsum[colB],   sB);
        atomicAdd(&colsumsq[colA], qA);
        atomicAdd(&colsumsq[colB], qB);
    }
}

// ---------------- BatchNorm finalize / apply ----------------
__global__ void bn_finalize_kernel(const float* __restrict__ colsum,
                                   const float* __restrict__ colsumsq, float invM,
                                   const float* __restrict__ g, const float* __restrict__ b,
                                   float* __restrict__ avec, float* __restrict__ cvec) {
    int j = threadIdx.x;                       // 256
    float m = colsum[j] * invM;
    float v = colsumsq[j] * invM - m * m;
    float a = g[j] * rsqrtf(v + 1e-5f);
    avec[j] = a;
    cvec[j] = b[j] - m * a;
}

// y = z*a[col]+c[col]; out = y; accum += w*y
__global__ void bn_apply_kernel(const float* __restrict__ z, const float* __restrict__ a,
                                const float* __restrict__ c, float* __restrict__ out,
                                float* __restrict__ accum, float w, long n) {
    long i = (long)blockIdx.x * blockDim.x + threadIdx.x;
    if (i < n) {
        int col = (int)(i & (HF - 1));
        float y = z[i] * a[col] + c[col];
        out[i]   = y;
        accum[i] += w * y;
    }
}

// ---------------- pair gather ----------------
__global__ void gather_kernel(const float* __restrict__ gout, const float* __restrict__ dout,
                              const int* __restrict__ bidx, float* __restrict__ xcat, int b0) {
    int r = blockIdx.x;             // row within chunk
    int f = threadIdx.x;            // 256
    int gb = b0 + r;
    int gi = bidx[2 * gb];
    int di = bidx[2 * gb + 1];
    xcat[(size_t)r * 512 + f]       = gout[(size_t)gi * HF + f];
    xcat[(size_t)r * 512 + HF + f]  = dout[(size_t)di * HF + f];
}

// ---------------- final 512 -> 2 linear ----------------
__global__ void final_kernel(const float* __restrict__ h, const float* __restrict__ W,
                             const float* __restrict__ b, float* __restrict__ out,
                             int rows, int b0) {
    int idx = blockIdx.x * blockDim.x + threadIdx.x;
    if (idx >= rows * 2) return;
    int r = idx >> 1, j = idx & 1;
    const float* hp = h + (size_t)r * 512;
    float s = b[j];
#pragma unroll 8
    for (int k = 0; k < 512; ++k) s += hp[k] * W[k * 2 + j];
    out[(size_t)(b0 + r) * 2 + j] = s;
}

// ---------------- host launch ----------------
extern "C" void kernel_launch(void* const* d_in, const int* in_sizes, int n_in,
                              void* d_out, int out_size, void* d_ws, size_t ws_size,
                              hipStream_t stream) {
    const float* gene_x    = (const float*)d_in[0];
    const float* disease_x = (const float*)d_in[1];
    const int*   g_ei      = (const int*)d_in[2];
    const int*   d_ei      = (const int*)d_in[3];
    const int*   batch_idx = (const int*)d_in[4];
    // per-layer weights: base 5 (gene) / 20 (disease), 5 entries each:
    // Wr, br, Ws, bn_g, bn_b
    const float* lin1_W = (const float*)d_in[35];
    const float* lin1_b = (const float*)d_in[36];
    const float* lin2_W = (const float*)d_in[37];
    const float* lin2_b = (const float*)d_in[38];
    const float* lin3_W = (const float*)d_in[39];
    const float* lin3_b = (const float*)d_in[40];
    const float* lin4_W = (const float*)d_in[41];
    const float* lin4_b = (const float*)d_in[42];

    float* ws = (float*)d_ws;
    size_t off = 0;
    auto alloc = [&](size_t n) { float* p = ws + off; off += (n + 63) & ~(size_t)63; return p; };
    float* agg    = alloc((size_t)NG * HF);
    float* zbuf   = alloc((size_t)NG * HF);
    float* ping   = alloc((size_t)NG * HF);
    float* pong   = alloc((size_t)NG * HF);
    float* gout   = alloc((size_t)NG * HF);
    float* doutb  = alloc((size_t)ND * HF);
    float* colsum   = alloc(256);
    float* colsumsq = alloc(256);
    float* avec     = alloc(256);
    float* cvec     = alloc(256);
    float* xcat = alloc((size_t)CHUNK * 512);
    float* h1   = alloc((size_t)CHUNK * 2048);
    float* h2   = alloc((size_t)CHUNK * 1024);
    float* h3   = alloc((size_t)CHUNK * 512);

    auto tower_layer = [&](const float* xin, int Nn, const int* ei, int E,
                           int widx, float* xout, float* accum, float w) {
        const float* Wr  = (const float*)d_in[widx + 0];
        const float* br  = (const float*)d_in[widx + 1];
        const float* Wsr = (const float*)d_in[widx + 2];
        const float* bg  = (const float*)d_in[widx + 3];
        const float* bb  = (const float*)d_in[widx + 4];
        long n = (long)Nn * HF;
        zero_kernel<<<(int)((n + 255) / 256), 256, 0, stream>>>(agg, n);
        scatter_add_kernel<<<E, 256, 0, stream>>>(xin, ei, E, agg);
        zero_kernel<<<2, 256, 0, stream>>>(colsum, 512);   // colsum+colsumsq contiguous
        dim3 grid((Nn + BM - 1) / BM, HF / BN);
        wmma_gemm_kernel<<<grid, 256, 0, stream>>>(agg, Wr, xin, Wsr, br, zbuf,
                                                   Nn, HF, HF, HF, 1, colsum, colsumsq);
        bn_finalize_kernel<<<1, 256, 0, stream>>>(colsum, colsumsq, 1.0f / (float)Nn,
                                                  bg, bb, avec, cvec);
        bn_apply_kernel<<<(int)((n + 255) / 256), 256, 0, stream>>>(zbuf, avec, cvec,
                                                                    xout, accum, w, n);
    };

    // ---- gene tower ----
    zero_kernel<<<(int)(((long)NG * HF + 255) / 256), 256, 0, stream>>>(gout, (long)NG * HF);
    tower_layer(gene_x, NG, g_ei, EG,  5, ping, gout, 0.7f);
    tower_layer(ping,   NG, g_ei, EG, 10, pong, gout, 0.2f);
    tower_layer(pong,   NG, g_ei, EG, 15, ping, gout, 0.1f);

    // ---- disease tower ----
    zero_kernel<<<(int)(((long)ND * HF + 255) / 256), 256, 0, stream>>>(doutb, (long)ND * HF);
    tower_layer(disease_x, ND, d_ei, ED, 20, ping, doutb, 0.7f);
    tower_layer(ping,      ND, d_ei, ED, 25, pong, doutb, 0.2f);
    tower_layer(pong,      ND, d_ei, ED, 30, ping, doutb, 0.1f);

    // ---- MLP head, chunked over batch rows ----
    for (int c0 = 0; c0 < BB; c0 += CHUNK) {
        gather_kernel<<<CHUNK, 256, 0, stream>>>(gout, doutb, batch_idx, xcat, c0);
        dim3 g1(CHUNK / BM, 2048 / BN);
        wmma_gemm_kernel<<<g1, 256, 0, stream>>>(xcat, lin1_W, nullptr, nullptr, lin1_b,
                                                 h1, CHUNK, 2048, 512, 0, 1, nullptr, nullptr);
        dim3 g2(CHUNK / BM, 1024 / BN);
        wmma_gemm_kernel<<<g2, 256, 0, stream>>>(h1, lin2_W, nullptr, nullptr, lin2_b,
                                                 h2, CHUNK, 1024, 2048, 0, 1, nullptr, nullptr);
        dim3 g3(CHUNK / BM, 512 / BN);
        wmma_gemm_kernel<<<g3, 256, 0, stream>>>(h2, lin3_W, nullptr, nullptr, lin3_b,
                                                 h3, CHUNK, 512, 1024, 0, 1, nullptr, nullptr);
        final_kernel<<<(CHUNK * 2 + 255) / 256, 256, 0, stream>>>(h3, lin4_W, lin4_b,
                                                                  (float*)d_out, CHUNK, c0);
    }
}
